// RNNModel_42949672960911
// MI455X (gfx1250) — compile-verified, matrix operand order checked by reading, not run on hardware
//
#include <hip/hip_runtime.h>
#include <hip/hip_bf16.h>

// ---- problem constants (match reference) ----
#define T_ 128
#define B_ 32
#define L_ 2
#define H_ 1024
#define E_ 1024
#define V_ 32000
#define G_ 4096   // 4*H

typedef __attribute__((ext_vector_type(16))) __bf16        v16bf;
typedef __attribute__((ext_vector_type(8)))  float         v8f;
typedef __attribute__((ext_vector_type(4))) unsigned int  u32x4;
typedef __attribute__((ext_vector_type(4)))  int           i32x4;
typedef __attribute__((ext_vector_type(8)))  int           i32x8;

// LDS row pitch for TDM-staged weight tiles: 32 bf16 (64B) + 16B pad = 80B.
// 16B-aligned for ds_load_b128; 20-dword bank stride is conflict-free mod 64.
#define BPITCH 40   // in ushorts

// round-to-nearest-even fp32 -> bf16 (as raw ushort)
static __device__ __forceinline__ unsigned short f2bf(float x) {
    union { float f; unsigned u; } v; v.f = x;
    unsigned r = v.u + 0x7FFFu + ((v.u >> 16) & 1u);
    return (unsigned short)(r >> 16);
}

// One lane's 16 bf16 elements of a 16x32 WMMA operand:
// two contiguous 16-byte chunks at p and p+16 elements (ISA 16-bit A/B layout).
static __device__ __forceinline__ v16bf load_bf16x16(const unsigned short* p) {
    struct Q { u32x4 a, b; } q;
    q.a = *reinterpret_cast<const u32x4*>(p);
    q.b = *reinterpret_cast<const u32x4*>(p + 16);
    return __builtin_bit_cast(v16bf, q);
}

static __device__ __forceinline__ float sigmoidf_(float x) {
    return 1.0f / (1.0f + __expf(-x));
}
static __device__ __forceinline__ float tanhf_(float x) {
    float ax = fabsf(x);
    float e  = __expf(-2.0f * ax);
    float t  = (1.0f - e) / (1.0f + e);
    return copysignf(t, x);
}

// ---- Tensor Data Mover: async-load a [nrows x 32] bf16 tile (row stride K elems)
// from global into LDS at lds_addr, padded +16B every 64B (row pitch 80B).
static __device__ __forceinline__ void tdm_load_tile(const unsigned short* gsrc,
                                                     unsigned lds_addr,
                                                     int K, int nrows) {
    unsigned long long ga = (unsigned long long)gsrc;
    u32x4 g0;
    g0.x = 1u;                                  // count=1, user mode, no gather
    g0.y = lds_addr;                            // LDS byte address
    g0.z = (unsigned)ga;                        // global_addr[31:0]
    g0.w = (unsigned)(ga >> 32) | (2u << 30);   // global_addr[56:32] | type=2

    i32x8 g1;
    g1[0] = (1 << 16)      // data_size = 2 bytes
          | (1 << 20)      // pad_enable
          | (3 << 22)      // pad_interval: every 16 dwords (64B)
          | (3 << 25);     // pad_amount: 4 dwords (16B)
    g1[1] = (K & 0xffff) << 16;                         // tensor_dim0[15:0]
    g1[2] = (K >> 16) & 0xffff;                         // tensor_dim0[31:16], td1 lo16=0
    g1[3] = 16 | (32 << 16);                            // td1 hi16 (td1=1<<20), tile_dim0=32
    g1[4] = nrows;                                      // tile_dim1, tile_dim2=0
    g1[5] = K;                                          // tensor_dim0_stride[31:0]
    g1[6] = 0;
    g1[7] = 0;

    i32x4 z4 = {};
#if defined(__clang_major__) && (__clang_major__ >= 23)
    i32x8 z8 = {};
    __builtin_amdgcn_tensor_load_to_lds(g0, g1, z4, z4, z8, 0);
#else
    __builtin_amdgcn_tensor_load_to_lds(g0, g1, z4, z4, 0);
#endif
}

// ---------------- elementwise helpers ----------------

__global__ void cvt_f32_bf16(const float* __restrict__ src,
                             unsigned short* __restrict__ dst, long long n) {
    long long base = (long long)blockIdx.x * 1024 + threadIdx.x;
#pragma unroll
    for (int i = 0; i < 4; ++i) {
        long long idx = base + (long long)i * 256;
        if (idx < n) dst[idx] = f2bf(src[idx]);
    }
}

__global__ void bias_sum(const float* __restrict__ a, const float* __restrict__ b,
                         float* __restrict__ o, int n) {
    int i = blockIdx.x * 256 + threadIdx.x;
    if (i < n) o[i] = a[i] + b[i];
}

__global__ void embed_kernel(const int* __restrict__ idx,
                             const float* __restrict__ emb,
                             unsigned short* __restrict__ x) {
    int tb = blockIdx.x;                       // 0..T*B-1
    long long row = idx[tb];
    const float* src = emb + row * (long long)E_;
    unsigned short* dst = x + (long long)tb * E_;
#pragma unroll
    for (int i = 0; i < 4; ++i) {
        int e = threadIdx.x + i * 256;
        dst[e] = f2bf(src[e]);
    }
}

// ---------------- bf16 WMMA GEMM: C[M,N] = A[M,K] * W[N,K]^T + bias[N] ----------------
// Block tile 128x64: 8 waves, wave w owns rows w*16..w*16+15 and all 64 cols
// (4 n-tiles). Weight K-slices (64 rows x 32 K) are TDM-staged into LDS,
// double-buffered; A fragments are register-double-buffered from global.

__global__ void gemm_bf16_wmma(const unsigned short* __restrict__ A,
                               const unsigned short* __restrict__ W,
                               const float* __restrict__ bias,
                               float* __restrict__ C,
                               int M, int N, int K) {
    __shared__ unsigned short bsm[2][64 * BPITCH];   // 2 x 5KB

    const int lane  = threadIdx.x & 31;
    const int wave  = threadIdx.x >> 5;
    const int halfk = (lane >> 4) << 3;              // 0 or 8
    const int lm    = lane & 15;

    const int mbase = blockIdx.y * 128 + wave * 16;
    const int nbase = blockIdx.x * 64;
    const int mrow  = mbase + lm;

    const unsigned lds0 = (unsigned)(unsigned long long)&bsm[0][0];
    const unsigned lds1 = (unsigned)(unsigned long long)&bsm[1][0];
    const unsigned short* wslice = W + (size_t)nbase * K;   // tile origin; +k0 along K

    v8f acc[4] = {};

    // prologue: stage first K-slice
    if (wave == 0) {
        tdm_load_tile(wslice, lds0, K, 64);
        __builtin_amdgcn_s_wait_tensorcnt(0);
    }
    __syncthreads();

    v16bf av = load_bf16x16(A + (size_t)mrow * K + halfk);

    int cur = 0;
    for (int k0 = 0; k0 < K; k0 += 32) {
        const bool more = (k0 + 32) < K;
        // kick off async TDM for next slice into the other buffer
        if (wave == 0 && more)
            tdm_load_tile(wslice + (k0 + 32), (cur ? lds0 : lds1), K, 64);

        // register-double-buffer A
        v16bf av_next = av;
        if (more)
            av_next = load_bf16x16(A + (size_t)mrow * K + (k0 + 32) + halfk);

        // four B fragments from the TDM-staged LDS tile
        const unsigned short* bbase = &bsm[cur][0] + (size_t)lm * BPITCH + halfk;
        v16bf bv0 = load_bf16x16(bbase + 0 * 16 * BPITCH);
        v16bf bv1 = load_bf16x16(bbase + 1 * 16 * BPITCH);
        v16bf bv2 = load_bf16x16(bbase + 2 * 16 * BPITCH);
        v16bf bv3 = load_bf16x16(bbase + 3 * 16 * BPITCH);

        acc[0] = __builtin_amdgcn_wmma_f32_16x16x32_bf16(
            false, av, false, bv0, (short)0, acc[0], false, false);
        acc[1] = __builtin_amdgcn_wmma_f32_16x16x32_bf16(
            false, av, false, bv1, (short)0, acc[1], false, false);
        acc[2] = __builtin_amdgcn_wmma_f32_16x16x32_bf16(
            false, av, false, bv2, (short)0, acc[2], false, false);
        acc[3] = __builtin_amdgcn_wmma_f32_16x16x32_bf16(
            false, av, false, bv3, (short)0, acc[3], false, false);

        if (wave == 0 && more)
            __builtin_amdgcn_s_wait_tensorcnt(0);
        __syncthreads();
        av  = av_next;
        cur ^= 1;
    }

#pragma unroll
    for (int j = 0; j < 4; ++j) {
        int c = nbase + j * 16 + lm;
        float bval = bias[c];
#pragma unroll
        for (int v = 0; v < 8; ++v) {
            int r = mbase + v + halfk;         // C/D layout: row = v + 8*(lane>=16)
            C[(size_t)r * N + c] = acc[j][v] + bval;
        }
    }
}

// ---------------- fused LSTM step: gates = xproj_t + h_prev * Whh^T, then gating ----------------
// grid = H/16 blocks (16 h-columns each); block = 256 threads = 8 waves:
// wave w -> mt = w&1 (batch half), gidx = w>>1 (i/f/g/o gate section).
// K-loop unrolling left to the compiler: deep unroll => many outstanding
// global_load_b128 pairs, which is what hides latency on the sequential path.

__global__ void lstm_step(const float* __restrict__ xproj_t,        // [B,4H]
                          const unsigned short* __restrict__ hprev, // [B,H] bf16
                          const unsigned short* __restrict__ Whh,   // [4H,H] bf16
                          const float* __restrict__ c_in,           // [B,H]
                          float* __restrict__ c_out,                // [B,H]
                          unsigned short* __restrict__ h_out,       // [B,H] bf16
                          float* __restrict__ h_final,              // [B,H] or null
                          float* __restrict__ c_final,              // [B,H] or null
                          int write_final) {
    __shared__ float lds[4][32][16];           // [gate][batch][h]

    const int lane  = threadIdx.x & 31;
    const int wave  = threadIdx.x >> 5;
    const int mt    = wave & 1;
    const int gidx  = wave >> 1;
    const int halfk = (lane >> 4) << 3;
    const int lm    = lane & 15;
    const int hs0   = blockIdx.x * 16;

    const int brow = mt * 16 + lm;             // batch row 0..31
    const int ncol = gidx * H_ + hs0 + lm;     // gate output column

    v8f acc = {};
    for (int k0 = 0; k0 < H_; k0 += 32) {
        v16bf av = load_bf16x16(hprev + (size_t)brow * H_ + k0 + halfk);
        v16bf bv = load_bf16x16(Whh + (size_t)ncol * H_ + k0 + halfk);
        acc = __builtin_amdgcn_wmma_f32_16x16x32_bf16(
            false, av, false, bv, (short)0, acc, false, false);
    }

#pragma unroll
    for (int v = 0; v < 8; ++v)
        lds[gidx][mt * 16 + v + halfk][lm] = acc[v];
    __syncthreads();

#pragma unroll
    for (int rep = 0; rep < 2; ++rep) {
        int e  = threadIdx.x + rep * 256;      // 0..511 -> (b, hh)
        int b  = e >> 4;
        int hh = e & 15;
        int col = hs0 + hh;
        float gi = lds[0][b][hh] + xproj_t[(size_t)b * G_ + 0 * H_ + col];
        float gf = lds[1][b][hh] + xproj_t[(size_t)b * G_ + 1 * H_ + col];
        float gg = lds[2][b][hh] + xproj_t[(size_t)b * G_ + 2 * H_ + col];
        float go = lds[3][b][hh] + xproj_t[(size_t)b * G_ + 3 * H_ + col];
        float i = sigmoidf_(gi);
        float f = sigmoidf_(gf);
        float g = tanhf_(gg);
        float o = sigmoidf_(go);
        float c = f * c_in[(size_t)b * H_ + col] + i * g;
        float h = o * tanhf_(c);
        c_out[(size_t)b * H_ + col] = c;
        h_out[(size_t)b * H_ + col] = f2bf(h);
        if (write_final) {
            h_final[(size_t)b * H_ + col] = h;
            c_final[(size_t)b * H_ + col] = c;
        }
    }
}

// ---------------- host orchestration ----------------

extern "C" void kernel_launch(void* const* d_in, const int* in_sizes, int n_in,
                              void* d_out, int out_size, void* d_ws, size_t ws_size,
                              hipStream_t stream) {
    const int*   input = (const int*)  d_in[0];
    const float* h0    = (const float*)d_in[1];
    const float* c0    = (const float*)d_in[2];
    const float* emb_w = (const float*)d_in[3];
    const float* W_ih  = (const float*)d_in[4];
    const float* W_hh  = (const float*)d_in[5];
    const float* b_ih  = (const float*)d_in[6];
    const float* b_hh  = (const float*)d_in[7];
    const float* dec_W = (const float*)d_in[8];
    const float* dec_b = (const float*)d_in[9];
    float* out = (float*)d_out;

    char*  ws  = (char*)d_ws;
    size_t off = 0;
    auto alloc = [&](size_t bytes) -> void* {
        void* p = ws + off;
        off += (bytes + 255) & ~(size_t)255;
        return p;
    };

    unsigned short* Wih_bf  = (unsigned short*)alloc((size_t)L_ * G_ * E_ * 2);
    unsigned short* Whh_bf  = (unsigned short*)alloc((size_t)L_ * G_ * H_ * 2);
    unsigned short* decW_bf = (unsigned short*)alloc((size_t)V_ * H_ * 2);
    float*          bias    = (float*)         alloc((size_t)L_ * G_ * 4);
    unsigned short* h0_bf   = (unsigned short*)alloc((size_t)L_ * B_ * H_ * 2);
    unsigned short* x_emb   = (unsigned short*)alloc((size_t)T_ * B_ * E_ * 2);
    unsigned short* out0    = (unsigned short*)alloc((size_t)T_ * B_ * H_ * 2);
    unsigned short* out1    = (unsigned short*)alloc((size_t)T_ * B_ * H_ * 2);
    float*          xproj   = (float*)         alloc((size_t)T_ * B_ * G_ * 4);
    float*          c_buf   = (float*)         alloc((size_t)B_ * H_ * 4);

    auto cvt = [&](const float* s, unsigned short* d, long long n) {
        int blocks = (int)((n + 1023) / 1024);
        cvt_f32_bf16<<<dim3(blocks), dim3(256), 0, stream>>>(s, d, n);
    };
    cvt(W_ih,  Wih_bf,  (long long)L_ * G_ * E_);
    cvt(W_hh,  Whh_bf,  (long long)L_ * G_ * H_);
    cvt(dec_W, decW_bf, (long long)V_ * H_);
    cvt(h0,    h0_bf,   (long long)L_ * B_ * H_);
    bias_sum<<<dim3((L_ * G_ + 255) / 256), dim3(256), 0, stream>>>(b_ih, b_hh, bias, L_ * G_);
    embed_kernel<<<dim3(T_ * B_), dim3(256), 0, stream>>>(input, emb_w, x_emb);

    for (int l = 0; l < L_; ++l) {
        const unsigned short* xin = (l == 0) ? x_emb : out0;
        unsigned short*       ob  = (l == 0) ? out0  : out1;

        // x_proj[t*B+b, g] = x . W_ih[l]^T + (b_ih+b_hh)   -- M=4096, N=4096, K=1024
        gemm_bf16_wmma<<<dim3(G_ / 64, (T_ * B_) / 128), dim3(256), 0, stream>>>(
            xin, Wih_bf + (size_t)l * G_ * E_, bias + (size_t)l * G_, xproj,
            T_ * B_, G_, E_);

        for (int t = 0; t < T_; ++t) {
            const unsigned short* hp = (t == 0) ? (h0_bf + (size_t)l * B_ * H_)
                                                : (ob + (size_t)(t - 1) * B_ * H_);
            const float* ci = (t == 0) ? (c0 + (size_t)l * B_ * H_) : c_buf;
            int wf = (t == T_ - 1) ? 1 : 0;
            float* hf = wf ? (out + (size_t)T_ * B_ * V_ + (size_t)l * B_ * H_) : nullptr;
            float* cf = wf ? (out + (size_t)T_ * B_ * V_ + (size_t)L_ * B_ * H_
                                  + (size_t)l * B_ * H_) : nullptr;
            lstm_step<<<dim3(H_ / 16), dim3(256), 0, stream>>>(
                xproj + (size_t)t * B_ * G_, hp, Whh_bf + (size_t)l * G_ * H_,
                ci, c_buf, ob + (size_t)t * B_ * H_, hf, cf, wf);
        }
    }

    // decoder: [4096,1024] x [32000,1024]^T + dec_b -> d_out (fp32)
    gemm_bf16_wmma<<<dim3(V_ / 64, (T_ * B_) / 128), dim3(256), 0, stream>>>(
        out1, decW_bf, dec_b, out, T_ * B_, V_, H_);
}
